// FCNN_BesselSine_41798621724974
// MI455X (gfx1250) — matrix operand order repaired and checked by reading, not compiled, not run on hardware
//
#include <hip/hip_runtime.h>
#include <hip/hip_bf16.h>

// ---------------------------------------------------------------------------
// Types for CDNA5 WMMA (wave32): v_wmma_f32_16x16x32_bf16
// ---------------------------------------------------------------------------
typedef __bf16 bf16_t;
typedef __attribute__((ext_vector_type(16))) __bf16 v16bf;
typedef __attribute__((ext_vector_type(8)))  __bf16 v8bf;
typedef __attribute__((ext_vector_type(8)))  float  v8f;

// exact pointee type of the async-copy builtins (per hipcc diagnostic)
typedef int v4i __attribute__((vector_size(16)));
typedef __attribute__((address_space(1))) v4i* gas_v4i_ptr;
typedef __attribute__((address_space(3))) v4i* las_v4i_ptr;

static __device__ __forceinline__ bf16_t f2bf(float f) {
    unsigned u = __float_as_uint(f);
    u += 0x7FFFu + ((u >> 16) & 1u);            // round-to-nearest-even
    unsigned short h = (unsigned short)(u >> 16);
    union { unsigned short s; bf16_t b; } cvt;
    cvt.s = h;
    return cvt.b;
}

// ---------------------------------------------------------------------------
// gfx1250 async global->LDS copy (no staging VGPRs, tracked by ASYNCcnt)
// ---------------------------------------------------------------------------
static __device__ __forceinline__ void async_copy_b128(void* lds, const void* g) {
#if __has_builtin(__builtin_amdgcn_global_load_async_to_lds_b128)
    __builtin_amdgcn_global_load_async_to_lds_b128(
        (gas_v4i_ptr)g, (las_v4i_ptr)lds, 0, 0);
#else
    *reinterpret_cast<uint4*>(lds) = *reinterpret_cast<const uint4*>(g);
#endif
}

template <int N>
static __device__ __forceinline__ void wait_asynccnt() {
#if __has_builtin(__builtin_amdgcn_s_wait_asynccnt)
    __builtin_amdgcn_s_wait_asynccnt(N);
#elif __has_builtin(__builtin_amdgcn_global_load_async_to_lds_b128)
    asm volatile("s_wait_asynccnt %0" ::"i"(N) : "memory");
#endif
}

// J1 Bessel (Numerical Recipes form, matches reference; fast-math transcendentals
// since the bf16 GEMM already bounds overall accuracy at ~3e-3)
static __device__ __forceinline__ float j1_approx(float x) {
    float ax = fabsf(x);
    if (ax < 8.0f) {
        float y = x * x;
        float num = x * (72362614232.0f + y * (-7895059235.0f + y * (242396853.1f
                  + y * (-2972611.439f + y * (15704.48260f + y * (-30.16036606f))))));
        float den = 144725228442.0f + y * (2300535178.0f + y * (18583304.74f
                  + y * (99447.43394f + y * (376.9991397f + y))));
        return __fdividef(num, den);
    } else {
        float z  = 8.0f / ax;
        float yy = z * z;
        float xx = ax - 2.356194491f;       // ax - 3*pi/4
        float p1 = 1.0f + yy * (0.183105e-2f + yy * (-0.3516396496e-4f
                 + yy * (0.2457520174e-5f + yy * (-0.240337019e-6f))));
        float p2 = 0.04687499995f + yy * (-0.2002690873e-3f + yy * (0.8449199096e-5f
                 + yy * (-0.88228987e-6f + yy * 0.105787412e-6f)));
        float s = __sinf(xx), c = __cosf(xx);
        float r = sqrtf(0.636619772f / ax) * (c * p1 - z * s * p2);
        return (x < 0.0f) ? -r : r;
    }
}

// ---------------------------------------------------------------------------
// Cast fp32 -> bf16 (elementwise)
// ---------------------------------------------------------------------------
__global__ void cast_f32_bf16(const float* __restrict__ in,
                              bf16_t* __restrict__ out, int n) {
    int i = blockIdx.x * blockDim.x + threadIdx.x;
    if (i < n) out[i] = f2bf(in[i]);
}

// Cast + transpose: in [R][C] fp32  ->  out [C][R] bf16   (block 32x8)
__global__ void cast_transpose_f32_bf16(const float* __restrict__ in,
                                        bf16_t* __restrict__ out,
                                        int R, int C) {
    __shared__ float tile[32][33];
    const int c0 = blockIdx.x * 32, r0 = blockIdx.y * 32;
    const int tx = threadIdx.x, ty = threadIdx.y;
    #pragma unroll
    for (int i = ty; i < 32; i += 8) {
        int r = r0 + i, c = c0 + tx;
        tile[i][tx] = (r < R && c < C) ? in[(size_t)r * C + c] : 0.0f;
    }
    __syncthreads();
    #pragma unroll
    for (int i = ty; i < 32; i += 8) {
        int r = r0 + tx, c = c0 + i;
        if (r < R && c < C) out[(size_t)c * R + r] = f2bf(tile[tx][i]);
    }
}

// ---------------------------------------------------------------------------
// Fused GEMM + bias + activation.
//   A  : [M][K]  bf16 row-major
//   Bt : [N][K]  bf16 (B pre-transposed => K-contiguous rows)
//   out: [M][N]  (bf16 or fp32)
// BM x BN block tile, BK=32, 256 threads = 8 wave32 waves in a WGM x WGN grid.
// Double-buffered LDS fed by GLOBAL_LOAD_ASYNC_TO_LDS_B128 (ASYNCcnt pipeline).
// ACT: 0 = J1 bessel, 1 = sin, 2 = identity
// ---------------------------------------------------------------------------
template <int BM, int BN, int WGM, int WGN, int ACT, bool OUTF32>
__global__ __launch_bounds__(256)
void gemm_bias_act(const bf16_t* __restrict__ A,
                   const bf16_t* __restrict__ Bt,
                   const float*  __restrict__ bias,
                   void* __restrict__ outv,
                   int M, int N, int K) {
    constexpr int BK  = 32;
    constexpr int BKP = BK + 8;            // pad to dodge LDS bank conflicts
    constexpr int WTM = BM / WGM;          // per-wave tile (rows)
    constexpr int WTN = BN / WGN;          // per-wave tile (cols)
    constexpr int MT  = WTM / 16;
    constexpr int NT  = WTN / 16;
    constexpr int ACH = (BM * BK) / (256 * 8);   // b128 chunks per thread (A)
    constexpr int BCH = (BN * BK) / (256 * 8);   // b128 chunks per thread (B)
    constexpr int CH  = ACH + BCH;               // async ops per thread per tile

    alignas(16) __shared__ bf16_t As[2][BM][BKP];
    alignas(16) __shared__ bf16_t Bs[2][BN][BKP];

    const int tid  = threadIdx.x;
    const int wid  = tid >> 5;
    const int lane = tid & 31;
    const int lh   = lane >> 4;        // half-wave select
    const int lm   = lane & 15;
    const int wm   = wid / WGN;
    const int wn   = wid % WGN;

    const int m0 = blockIdx.x * BM;
    const int n0 = blockIdx.y * BN;

    v8f acc[MT][NT];
    #pragma unroll
    for (int i = 0; i < MT; ++i)
        #pragma unroll
        for (int j = 0; j < NT; ++j)
            acc[i][j] = {};

    const int KT = K / BK;

    // direct global -> LDS async copy of one BK-tile into buffer `buf`
    auto issue = [&](int kt, int buf) {
        #pragma unroll
        for (int s = 0; s < ACH; ++s) {
            int c = tid + s * 256;
            int row = c >> 2, off = (c & 3) * 8;
            async_copy_b128(&As[buf][row][off],
                A + (size_t)(m0 + row) * K + (size_t)kt * BK + off);
        }
        #pragma unroll
        for (int s = 0; s < BCH; ++s) {
            int c = tid + s * 256;
            int row = c >> 2, off = (c & 3) * 8;
            async_copy_b128(&Bs[buf][row][off],
                Bt + (size_t)(n0 + row) * K + (size_t)kt * BK + off);
        }
    };

    auto compute = [&](int buf) {
        v16bf afrag[MT], bfrag[NT];
        #pragma unroll
        for (int i = 0; i < MT; ++i) {
            int row = wm * WTM + i * 16 + lm;
            v8bf alo = *reinterpret_cast<const v8bf*>(&As[buf][row][8 * lh]);
            v8bf ahi = *reinterpret_cast<const v8bf*>(&As[buf][row][16 + 8 * lh]);
            afrag[i] = __builtin_shufflevector(alo, ahi,
                0, 1, 2, 3, 4, 5, 6, 7, 8, 9, 10, 11, 12, 13, 14, 15);
        }
        #pragma unroll
        for (int j = 0; j < NT; ++j) {
            int n = wn * WTN + j * 16 + lm;
            v8bf blo = *reinterpret_cast<const v8bf*>(&Bs[buf][n][8 * lh]);
            v8bf bhi = *reinterpret_cast<const v8bf*>(&Bs[buf][n][16 + 8 * lh]);
            bfrag[j] = __builtin_shufflevector(blo, bhi,
                0, 1, 2, 3, 4, 5, 6, 7, 8, 9, 10, 11, 12, 13, 14, 15);
        }
        #pragma unroll
        for (int i = 0; i < MT; ++i)
            #pragma unroll
            for (int j = 0; j < NT; ++j)
                acc[i][j] = __builtin_amdgcn_wmma_f32_16x16x32_bf16(
                    false, afrag[i], false, bfrag[j],
                    (short)0, acc[i][j], false, false);
    };

    // software pipeline: tile kt+1 streams into LDS[buf^1] while computing LDS[buf]
    issue(0, 0);
    int buf = 0;
    for (int kt = 0; kt < KT; ++kt) {
        if (kt + 1 < KT) {
            issue(kt + 1, buf ^ 1);            // ASYNCcnt += CH
            if (kt + 2 < KT)                   // gfx1250 global_prefetch_b8
                __builtin_prefetch(A + (size_t)(m0 + (tid >> 2)) * K
                                     + (size_t)(kt + 2) * BK, 0, 3);
            wait_asynccnt<CH>();               // in-order => tile kt complete
        } else {
            wait_asynccnt<0>();
        }
        __syncthreads();                       // all waves' tile-kt writes visible
        compute(buf);
        __syncthreads();                       // reads of buf done before reuse
        buf ^= 1;
    }

    // epilogue: bias + activation + store
    #pragma unroll
    for (int j = 0; j < NT; ++j) {
        int col = n0 + wn * WTN + j * 16 + lm;
        float bv = bias[col];
        #pragma unroll
        for (int i = 0; i < MT; ++i) {
            int rbase = m0 + wm * WTM + i * 16 + 8 * lh;
            #pragma unroll
            for (int r = 0; r < 8; ++r) {
                float v = acc[i][j][r] + bv;
                if constexpr (ACT == 0)      v = j1_approx(v);
                else if constexpr (ACT == 1) v = __sinf(v);
                size_t o = (size_t)(rbase + r) * N + col;
                if constexpr (OUTF32) ((float*)outv)[o]  = v;
                else                  ((bf16_t*)outv)[o] = f2bf(v);
            }
        }
    }
}

// ---------------------------------------------------------------------------
// Launch: t(32768x64) -> J1(tW1+b1) -> sin(hW2+b2) -> hW3+b3
// ---------------------------------------------------------------------------
extern "C" void kernel_launch(void* const* d_in, const int* in_sizes, int n_in,
                              void* d_out, int out_size, void* d_ws, size_t ws_size,
                              hipStream_t stream) {
    (void)in_sizes; (void)n_in; (void)out_size; (void)ws_size;

    constexpr int Bm = 32768, Din = 64, H = 2048, Dout = 64;

    const float* t  = (const float*)d_in[0];
    const float* W1 = (const float*)d_in[1];
    const float* b1 = (const float*)d_in[2];
    const float* W2 = (const float*)d_in[3];
    const float* b2 = (const float*)d_in[4];
    const float* W3 = (const float*)d_in[5];
    const float* b3 = (const float*)d_in[6];
    float* out = (float*)d_out;

    // workspace layout (bf16), all offsets 256KB-multiples -> 16B aligned
    char* w = (char*)d_ws;
    bf16_t* tB  = (bf16_t*)w;  w += (size_t)Bm  * Din * 2;   //   4 MB
    bf16_t* W1T = (bf16_t*)w;  w += (size_t)H   * Din * 2;   // 256 KB  [H][Din]
    bf16_t* W2T = (bf16_t*)w;  w += (size_t)H   * H   * 2;   //   8 MB  [H][H]
    bf16_t* W3T = (bf16_t*)w;  w += (size_t)Dout* H   * 2;   // 256 KB  [Dout][H]
    bf16_t* h1  = (bf16_t*)w;  w += (size_t)Bm  * H   * 2;   // 128 MB
    bf16_t* h2  = (bf16_t*)w;                                // 128 MB

    // one-time casts / weight transposes (bf16, K-contiguous B operands)
    cast_f32_bf16<<<(Bm * Din + 255) / 256, 256, 0, stream>>>(t, tB, Bm * Din);
    dim3 tb(32, 8);
    cast_transpose_f32_bf16<<<dim3(H / 32,   Din / 32), tb, 0, stream>>>(W1, W1T, Din, H);
    cast_transpose_f32_bf16<<<dim3(H / 32,   H   / 32), tb, 0, stream>>>(W2, W2T, H,   H);
    cast_transpose_f32_bf16<<<dim3(Dout / 32, H  / 32), tb, 0, stream>>>(W3, W3T, H, Dout);

    // L1: [32768x64] x [64x2048]   + b1 -> J1        (bf16 out)
    gemm_bias_act<128, 128, 2, 4, 0, false>
        <<<dim3(Bm / 128, H / 128), 256, 0, stream>>>(tB, W1T, b1, h1, Bm, H, Din);
    // L2: [32768x2048] x [2048x2048] + b2 -> sin     (bf16 out)
    gemm_bias_act<128, 128, 2, 4, 1, false>
        <<<dim3(Bm / 128, H / 128), 256, 0, stream>>>(h1, W2T, b2, h2, Bm, H, H);
    // L3: [32768x2048] x [2048x64]  + b3 -> identity (fp32 out)
    gemm_bias_act<128, 64, 4, 2, 2, true>
        <<<dim3(Bm / 128, Dout / 64), 256, 0, stream>>>(h2, W3T, b3, out, Bm, Dout, H);
}
